// FilterImage_47665547051744
// MI455X (gfx1250) — compile-verified, matrix operand order checked by reading, not run on hardware
//
#include <hip/hip_runtime.h>
#include <hip/hip_bf16.h>

typedef __attribute__((ext_vector_type(16))) _Float16 v16h;
typedef __attribute__((ext_vector_type(8)))  float    v8f;

#define H_FULL 5000
#define W_FULL 4000
#define H_SM   1000
#define W_SM   800
#define POS_ROWS 1024   // zero-padded rows for WMMA K loops (col sums)
#define YIDX_N   1008   // 63 tiles * 16 rows

// ---------------------------------------------------------------------------
// Stage 1: fused 5x antialiased bilinear downsample (separable triangle,
// center = 5r+2 / 5c+2, taps d=-4..4 weight (5-|d|), renormalized per axis
// over in-bounds taps — matches jax.image.resize(antialias=True)).
// ---------------------------------------------------------------------------
__global__ __launch_bounds__(256) void downsample_kernel(const float* __restrict__ x,
                                                         float* __restrict__ x1) {
    int idx = blockIdx.x * blockDim.x + threadIdx.x;
    if (idx >= H_SM * W_SM) return;
    int r = idx / W_SM, c = idx % W_SM;
    int cy = 5 * r + 2, cx = 5 * c + 2;
    float wy[9], wx[9];
    float sy = 0.0f, sx = 0.0f;
    #pragma unroll
    for (int d = -4; d <= 4; ++d) {
        float u = 5.0f - fabsf((float)d);
        int ky = cy + d, kx = cx + d;
        float uy = (ky >= 0 && ky < H_FULL) ? u : 0.0f;
        float ux = (kx >= 0 && kx < W_FULL) ? u : 0.0f;
        wy[d + 4] = uy; sy += uy;
        wx[d + 4] = ux; sx += ux;
    }
    float acc = 0.0f;
    #pragma unroll
    for (int dy = 0; dy < 9; ++dy) {
        if (wy[dy] == 0.0f) continue;
        int ky = cy + dy - 4;
        const float* row = x + (size_t)ky * W_FULL;
        float racc = 0.0f;
        #pragma unroll
        for (int dx = 0; dx < 9; ++dx) {
            if (wx[dx] == 0.0f) continue;
            racc += wx[dx] * row[cx + dx - 4];
        }
        acc += wy[dy] * racc;
    }
    x1[idx] = acc / (sy * sx);
}

// ---------------------------------------------------------------------------
// Stage 2: separable 13-wide windowed min/max with zero padding (OOB -> 0).
// ---------------------------------------------------------------------------
template <bool HORIZ, bool ISMIN>
__global__ __launch_bounds__(256) void morph_kernel(const float* __restrict__ in,
                                                    float* __restrict__ out) {
    int idx = blockIdx.x * blockDim.x + threadIdx.x;
    if (idx >= H_SM * W_SM) return;
    int r = idx / W_SM, c = idx % W_SM;
    float best = ISMIN ? 3.402823e38f : -3.402823e38f;
    #pragma unroll
    for (int d = -6; d <= 6; ++d) {
        int rr = HORIZ ? r : r + d;
        int cc = HORIZ ? c + d : c;
        float v = (rr >= 0 && rr < H_SM && cc >= 0 && cc < W_SM)
                      ? in[rr * W_SM + cc] : 0.0f;
        best = ISMIN ? fminf(best, v) : fmaxf(best, v);
    }
    out[idx] = best;
}

// ---------------------------------------------------------------------------
// Stage 3: pos = (di > 0) as f16 0/1, zero-padded to 1024 rows.
// ---------------------------------------------------------------------------
__global__ __launch_bounds__(256) void pos_kernel(const float* __restrict__ di,
                                                  _Float16* __restrict__ pos) {
    int idx = blockIdx.x * blockDim.x + threadIdx.x;
    if (idx >= POS_ROWS * W_SM) return;
    int r = idx / W_SM;
    float v = (r < H_SM) ? di[idx] : 0.0f;
    pos[idx] = (v > 0.0f) ? (_Float16)1.0f : (_Float16)0.0f;
}

// ---------------------------------------------------------------------------
// Stage 4a: per-row counts via WMMA:  y_idx = pos(16x32 tiles) x ones(32x16).
// One wave per 16-row tile. A operand uses the ISA 16-bit 16x32 layout:
// lanes 0-15 hold K 0-7 & 16-23, lanes 16-31 hold K 8-15 & 24-31 for row M.
// D column 0 lives in lane 0 (M=0..7) and lane 16 (M=8..15).
// ---------------------------------------------------------------------------
__global__ __launch_bounds__(32) void rowsum_wmma_kernel(const _Float16* __restrict__ pos,
                                                         float* __restrict__ y_idx) {
    int tile = blockIdx.x;          // 63 tiles -> rows [0, 1008)
    int lane = threadIdx.x;
    int m = lane & 15;
    int row = tile * 16 + m;
    int khalf = (lane < 16) ? 0 : 8;

    v16h ones;
    #pragma unroll
    for (int i = 0; i < 16; ++i) ones[i] = (_Float16)1.0f;

    v8f c = {};
    for (int k0 = 0; k0 < W_SM; k0 += 32) {
        const _Float16* base = pos + (size_t)row * W_SM + k0;
        v16h a;
        #pragma unroll
        for (int j = 0; j < 8; ++j) {
            a[j]     = base[khalf + j];        // K = khalf + j
            a[8 + j] = base[16 + khalf + j];   // K = 16 + khalf + j
        }
        c = __builtin_amdgcn_wmma_f32_16x16x32_f16(
                false, a, false, ones, (short)0, c, false, false);
    }
    if (lane == 0) {
        #pragma unroll
        for (int i = 0; i < 8; ++i) y_idx[tile * 16 + i] = c[i];
    }
    if (lane == 16) {
        #pragma unroll
        for (int i = 0; i < 8; ++i) y_idx[tile * 16 + 8 + i] = c[i];
    }
}

// ---------------------------------------------------------------------------
// Stage 4b: per-col counts via WMMA: A = pos^T tile (transposed gather),
// B = ones. D[m,0] = sum_k pos[k, c0+m].  One wave per 16-column tile.
// K runs over 1024 zero-padded rows -> no tail handling.
// ---------------------------------------------------------------------------
__global__ __launch_bounds__(32) void colsum_wmma_kernel(const _Float16* __restrict__ pos,
                                                         float* __restrict__ x_idx) {
    int tile = blockIdx.x;          // 50 tiles -> cols [0, 800)
    int lane = threadIdx.x;
    int m = lane & 15;
    int col = tile * 16 + m;
    int khalf = (lane < 16) ? 0 : 8;

    v16h ones;
    #pragma unroll
    for (int i = 0; i < 16; ++i) ones[i] = (_Float16)1.0f;

    v8f c = {};
    for (int k0 = 0; k0 < POS_ROWS; k0 += 32) {
        v16h a;
        #pragma unroll
        for (int j = 0; j < 8; ++j) {
            a[j]     = pos[(size_t)(k0 + khalf + j) * W_SM + col];
            a[8 + j] = pos[(size_t)(k0 + 16 + khalf + j) * W_SM + col];
        }
        c = __builtin_amdgcn_wmma_f32_16x16x32_f16(
                false, a, false, ones, (short)0, c, false, false);
    }
    if (lane == 0) {
        #pragma unroll
        for (int i = 0; i < 8; ++i) x_idx[tile * 16 + i] = c[i];
    }
    if (lane == 16) {
        #pragma unroll
        for (int i = 0; i < 8; ++i) x_idx[tile * 16 + 8 + i] = c[i];
    }
}

// ---------------------------------------------------------------------------
// Stage 5: 1D linear upsample (align_corners=False) of counts, threshold > 0.
// ---------------------------------------------------------------------------
__global__ __launch_bounds__(256) void mask_kernel(const float* __restrict__ vals,
                                                   unsigned char* __restrict__ mask,
                                                   int N, int n) {
    int i = blockIdx.x * blockDim.x + threadIdx.x;
    if (i >= N) return;
    float src = (i + 0.5f) * ((float)n / (float)N) - 0.5f;
    src = fminf(fmaxf(src, 0.0f), (float)(n - 1));
    int i0 = (int)floorf(src);
    int i1 = min(i0 + 1, n - 1);
    float f = src - (float)i0;
    float v = (1.0f - f) * vals[i0] + f * vals[i1];
    mask[i] = (v > 0.0f) ? 1 : 0;
}

// ---------------------------------------------------------------------------
// Stage 6: streaming masked copy, float4 vectorized (the 160 MB stage).
// ---------------------------------------------------------------------------
__global__ __launch_bounds__(256) void final_mask_kernel(const float4* __restrict__ x,
                                                         float4* __restrict__ out,
                                                         const unsigned char* __restrict__ rm,
                                                         const unsigned char* __restrict__ cm) {
    const int W4 = W_FULL / 4;
    int idx = blockIdx.x * blockDim.x + threadIdx.x;
    if (idx >= H_FULL * W4) return;
    int r = idx / W4;
    int c = (idx % W4) * 4;
    float4 v = x[idx];
    unsigned char rmk = rm[r];
    float4 o;
    o.x = (rmk && cm[c + 0]) ? v.x : 0.0f;
    o.y = (rmk && cm[c + 1]) ? v.y : 0.0f;
    o.z = (rmk && cm[c + 2]) ? v.z : 0.0f;
    o.w = (rmk && cm[c + 3]) ? v.w : 0.0f;
    out[idx] = o;
}

extern "C" void kernel_launch(void* const* d_in, const int* in_sizes, int n_in,
                              void* d_out, int out_size, void* d_ws, size_t ws_size,
                              hipStream_t stream) {
    const float* x = (const float*)d_in[0];
    float* out = (float*)d_out;
    char* ws = (char*)d_ws;

    // Workspace layout (bytes), all 256-aligned:
    const size_t SM_BYTES = (size_t)H_SM * W_SM * sizeof(float);      // 3,200,000
    float*         bufA     = (float*)(ws);                            // x1 / er / di
    float*         bufB     = (float*)(ws + SM_BYTES);                 // intermediates
    _Float16*      pos      = (_Float16*)(ws + 2 * SM_BYTES);          // 1024x800 f16
    const size_t   POS_B    = (size_t)POS_ROWS * W_SM * sizeof(_Float16); // 1,638,400
    float*         y_idx    = (float*)(ws + 2 * SM_BYTES + POS_B);     // 1008 floats
    float*         x_idx    = (float*)(ws + 2 * SM_BYTES + POS_B + 4352);
    unsigned char* row_mask = (unsigned char*)(ws + 2 * SM_BYTES + POS_B + 4352 + 3328);
    unsigned char* col_mask = row_mask + 5120;

    const int sm_elems  = H_SM * W_SM;
    const int sm_blocks = (sm_elems + 255) / 256;

    // 1) antialiased 5x downsample: x (5000x4000) -> bufA (1000x800)
    downsample_kernel<<<sm_blocks, 256, 0, stream>>>(x, bufA);

    // 2) opening: erosion (min, zero pad) then dilation (max, zero pad), separable
    morph_kernel<true,  true ><<<sm_blocks, 256, 0, stream>>>(bufA, bufB); // min horiz
    morph_kernel<false, true ><<<sm_blocks, 256, 0, stream>>>(bufB, bufA); // min vert -> er
    morph_kernel<true,  false><<<sm_blocks, 256, 0, stream>>>(bufA, bufB); // max horiz
    morph_kernel<false, false><<<sm_blocks, 256, 0, stream>>>(bufB, bufA); // max vert -> di

    // 3) pos = (di > 0) as f16 0/1, zero-padded to 1024 rows
    pos_kernel<<<(POS_ROWS * W_SM + 255) / 256, 256, 0, stream>>>(bufA, pos);

    // 4) row/col positive counts via V_WMMA_F32_16X16X32_F16 (B = ones)
    rowsum_wmma_kernel<<<YIDX_N / 16, 32, 0, stream>>>(pos, y_idx);   // 63 waves
    colsum_wmma_kernel<<<W_SM / 16,  32, 0, stream>>>(pos, x_idx);    // 50 waves

    // 5) upsample counts to full res, threshold -> byte masks
    mask_kernel<<<(H_FULL + 255) / 256, 256, 0, stream>>>(y_idx, row_mask, H_FULL, H_SM);
    mask_kernel<<<(W_FULL + 255) / 256, 256, 0, stream>>>(x_idx, col_mask, W_FULL, W_SM);

    // 6) streaming masked copy (bandwidth-bound stage)
    const int n4 = H_FULL * (W_FULL / 4);
    final_mask_kernel<<<(n4 + 255) / 256, 256, 0, stream>>>(
        (const float4*)x, (float4*)out, row_mask, col_mask);
}
